// Trellis2_34600256537188
// MI455X (gfx1250) — matrix operand order repaired and checked by reading, compiled, not run on hardware
//
#include <hip/hip_runtime.h>
#include <hip/hip_bf16.h>

#define RDIM 32
#define NH   16
#define CD   1024
#define HD   64
#define NTOK (RDIM*RDIM*RDIM)   // 32768
#define NWIN 64
#define TWIN 512                // 8^3 tokens per window
#define FREQD 10
#define SM_SCALE 0.125f         // 1/sqrt(64)

typedef __bf16 bf16;
typedef __attribute__((ext_vector_type(16))) __bf16 v16bf;
typedef __attribute__((ext_vector_type(8)))  __bf16 v8bf;
typedef __attribute__((ext_vector_type(4)))  __bf16 v4bf;
typedef __attribute__((ext_vector_type(2)))  __bf16 v2bf;
typedef __attribute__((ext_vector_type(8)))  float  v8f;
typedef __attribute__((ext_vector_type(4)))  unsigned int v4u;
typedef __attribute__((ext_vector_type(8)))  int v8i;
typedef __attribute__((ext_vector_type(4)))  int v4i;

#if defined(__has_builtin)
#if __has_builtin(__builtin_amdgcn_tensor_load_to_lds)
#define HAVE_TDM 1
#endif
#endif

// Build a 16-element bf16 A/B fragment from two contiguous 8-element chunks.
static __device__ __forceinline__ v16bf ld16(const bf16* lo, const bf16* hi) {
  v8bf a = *(const v8bf*)lo;
  v8bf b = *(const v8bf*)hi;
  v16bf r;
#pragma unroll
  for (int i = 0; i < 8; ++i) { r[i] = a[i]; r[i + 8] = b[i]; }
  return r;
}

static __device__ __forceinline__ void zero8(v8f& v) {
#pragma unroll
  for (int i = 0; i < 8; ++i) v[i] = 0.0f;
}

#ifdef HAVE_TDM
// 2D tile DMA (global -> LDS) via the Tensor Data Mover.
// data_size = 2 bytes; tile_d0 elements per row, tile_d1 rows, row stride in elements.
// Tensor dims equal the tile so the OOB path never triggers. ISA 08 §8.3/8.4.
static __device__ __forceinline__ void tdm_load_2d(unsigned lds_off,
                                                   unsigned long long gaddr,
                                                   unsigned tile_d0, unsigned tile_d1,
                                                   unsigned stride_elems) {
  v4u g0;
  g0[0] = 1u;                                                  // count=1 valid user D#
  g0[1] = lds_off;                                             // lds_addr (bytes)
  g0[2] = (unsigned)(gaddr & 0xFFFFFFFFull);                   // global_addr[31:0]
  g0[3] = (unsigned)((gaddr >> 32) & 0x1FFFFFFull) | (2u << 30); // addr[56:32] | type=2
  v8i g1;
  g1[0] = (int)(1u << 16);                                     // data_size=1 -> 2 bytes
  g1[1] = (int)((tile_d0 & 0xFFFFu) << 16);                    // tensor_dim0[15:0]
  g1[2] = (int)(((tile_d0 >> 16) & 0xFFFFu) | ((tile_d1 & 0xFFFFu) << 16)); // td0 hi | td1 lo
  g1[3] = (int)(((tile_d1 >> 16) & 0xFFFFu) | ((tile_d0 & 0xFFFFu) << 16)); // td1 hi | tile_dim0
  g1[4] = (int)(tile_d1 & 0xFFFFu);                            // tile_dim1 (tile_dim2=0)
  g1[5] = (int)stride_elems;                                   // tensor_dim0_stride[31:0]
  g1[6] = 0;
  g1[7] = 0;
  v4i z4 = {0, 0, 0, 0};
#if __clang_major__ >= 23
  v8i z8 = {0, 0, 0, 0, 0, 0, 0, 0};
  __builtin_amdgcn_tensor_load_to_lds(g0, g1, z4, z4, z8, 0);
#else
  __builtin_amdgcn_tensor_load_to_lds(g0, g1, z4, z4, 0);
#endif
}
static __device__ __forceinline__ unsigned lds_off_of(const void* p) {
  return (unsigned)(unsigned long long)(uintptr_t)p;   // low 32 bits = LDS byte offset
}
#endif

// ---------------------------------------------------------------- fp32 -> bf16
__global__ __launch_bounds__(256) void cvt_bf16_kernel(const float* __restrict__ src,
                                                       bf16* __restrict__ dst, int n) {
  int i = (blockIdx.x * 256 + threadIdx.x) * 4;
  if (i >= n) return;
  float4 f = *(const float4*)(src + i);
  v4bf o;
  o[0] = (bf16)f.x; o[1] = (bf16)f.y; o[2] = (bf16)f.z; o[3] = (bf16)f.w;
  *(v4bf*)(dst + i) = o;
}

// ------------------------------------------- fp32 [K][N] -> bf16 [N][K] (weights)
// One-time transpose so GEMM B tiles become plain 2D copies for the TDM.
__global__ __launch_bounds__(256) void cvt_transpose_kernel(const float* __restrict__ src,
                                                            bf16* __restrict__ dst,
                                                            int K, int N) {
  int idx = blockIdx.x * 256 + threadIdx.x;   // over N * K/8
  int n = idx / (K / 8);
  int k8 = (idx % (K / 8)) * 8;
  if (n >= N) return;
  v8bf o;
#pragma unroll
  for (int j = 0; j < 8; ++j) o[j] = (bf16)src[(size_t)(k8 + j) * N + n];
  *(v8bf*)(dst + (size_t)n * K + k8) = o;
}

// --------------------------------------------- QKV GEMM + RMSNorm + RoPE + window
// grid: (48 column blocks of 64 = one head of q|k|v, 256 row tiles of 128), 256 thr
// layouts out: qw2/kw2 [(w*NH+h)][t][d], vwT [(w*NH+h)][d][t]
__global__ __launch_bounds__(256) void qkv_gemm_kernel(
    const bf16* __restrict__ xb, const bf16* __restrict__ wqbT,
    const float* __restrict__ bqkv,
    const float* __restrict__ gq, const float* __restrict__ gk,
    bf16* __restrict__ qw2, bf16* __restrict__ kw2, bf16* __restrict__ vwT) {
  __shared__ __align__(16) bf16 aBuf[2][128 * 64];   // [m][k]  2x16KB
  __shared__ __align__(16) bf16 bBuf[2][64 * 64];    // [n][k]  2x8KB
  __shared__ __align__(16) float cLds[128 * 64];     // 32KB

  const int b = blockIdx.x;          // 0..47
  const int rowbase = blockIdx.y * 128;
  const int tid = threadIdx.x, wave = tid >> 5, lane = tid & 31;
  const int s = b >> 4;              // 0=q 1=k 2=v
  const int h = b & 15;
  const int ncol = b * 64;
  const int NITER = CD / 64;

  const bf16* aSrc = xb + (size_t)rowbase * CD;        // + kk
  const bf16* bSrc = wqbT + (size_t)ncol * CD;         // + kk, row stride CD

  v8f acc[4];
#pragma unroll
  for (int i = 0; i < 4; ++i) zero8(acc[i]);

  // prologue: stage tile 0
#ifdef HAVE_TDM
  if (tid == 0) {
    tdm_load_2d(lds_off_of(&aBuf[0][0]), (unsigned long long)(uintptr_t)aSrc, 64, 128, CD);
    tdm_load_2d(lds_off_of(&bBuf[0][0]), (unsigned long long)(uintptr_t)bSrc, 64, 64, CD);
    __builtin_amdgcn_s_wait_tensorcnt((short)0);
  }
#else
  {
    int row = tid >> 1, half = tid & 1;
#pragma unroll
    for (int j = 0; j < 4; ++j)
      *(v8bf*)&aBuf[0][row * 64 + half * 32 + 8 * j] =
          *(const v8bf*)(aSrc + (size_t)row * CD + half * 32 + 8 * j);
    int n = tid >> 2, q = (tid & 3) * 16;
    *(v8bf*)&bBuf[0][n * 64 + q]     = *(const v8bf*)(bSrc + (size_t)n * CD + q);
    *(v8bf*)&bBuf[0][n * 64 + q + 8] = *(const v8bf*)(bSrc + (size_t)n * CD + q + 8);
  }
#endif
  __syncthreads();

  for (int it = 0; it < NITER; ++it) {
    const int cur = it & 1, nxt = cur ^ 1;
    const int kk = (it + 1) * 64;
    if (it + 1 < NITER) {   // stage next tile while this one computes
#ifdef HAVE_TDM
      if (tid == 0) {
        tdm_load_2d(lds_off_of(&aBuf[nxt][0]),
                    (unsigned long long)(uintptr_t)(aSrc + kk), 64, 128, CD);
        tdm_load_2d(lds_off_of(&bBuf[nxt][0]),
                    (unsigned long long)(uintptr_t)(bSrc + kk), 64, 64, CD);
      }
#else
      int row = tid >> 1, half = tid & 1;
#pragma unroll
      for (int j = 0; j < 4; ++j)
        *(v8bf*)&aBuf[nxt][row * 64 + half * 32 + 8 * j] =
            *(const v8bf*)(aSrc + (size_t)row * CD + kk + half * 32 + 8 * j);
      int n = tid >> 2, q = (tid & 3) * 16;
      *(v8bf*)&bBuf[nxt][n * 64 + q]     = *(const v8bf*)(bSrc + (size_t)n * CD + kk + q);
      *(v8bf*)&bBuf[nxt][n * 64 + q + 8] = *(const v8bf*)(bSrc + (size_t)n * CD + kk + q + 8);
#endif
    }
    const bf16* aC = aBuf[cur];
    const bf16* bC = bBuf[cur];
    const int m = wave * 16 + (lane & 15);
    const int kb8 = (lane >> 4) * 8;
    const int kb16 = (lane >> 4) * 16;
    v16bf af0 = ld16(&aC[m * 64 + kb16],      &aC[m * 64 + kb16 + 8]);
    v16bf af1 = ld16(&aC[m * 64 + 32 + kb16], &aC[m * 64 + 32 + kb16 + 8]);
    v16bf bf0[4], bf1[4];
#pragma unroll
    for (int nt = 0; nt < 4; ++nt) {
      const bf16* bp = &bC[(nt * 16 + (lane & 15)) * 64 + kb16];
      bf0[nt] = *(const v16bf*)bp;
      bf1[nt] = *(const v16bf*)(bp + 32);
    }
#pragma unroll
    for (int nt = 0; nt < 4; ++nt)
      acc[nt] = __builtin_amdgcn_wmma_f32_16x16x32_bf16(false, af0, false, bf0[nt],
                                                        (short)0, acc[nt], false, false);
#pragma unroll
    for (int nt = 0; nt < 4; ++nt)
      acc[nt] = __builtin_amdgcn_wmma_f32_16x16x32_bf16(false, af1, false, bf1[nt],
                                                        (short)0, acc[nt], false, false);
#ifdef HAVE_TDM
    if (tid == 0) __builtin_amdgcn_s_wait_tensorcnt((short)0);
#endif
    __syncthreads();
  }

#pragma unroll
  for (int nt = 0; nt < 4; ++nt)
#pragma unroll
    for (int r = 0; r < 8; ++r)
      cLds[(wave * 16 + r + 8 * (lane >> 4)) * 64 + nt * 16 + (lane & 15)] = acc[nt][r];
  // epilogue is intra-wave on the rows this wave produced (LDS in-order per wave)
  const float* gam = (s == 0) ? gq : gk;
#pragma unroll 1
  for (int r = 0; r < 16; ++r) {
    const int m = wave * 16 + r;
    const int n = rowbase + m;
    float c0 = cLds[m * 64 + 2 * lane]     + bqkv[ncol + 2 * lane];
    float c1 = cLds[m * 64 + 2 * lane + 1] + bqkv[ncol + 2 * lane + 1];
    const int gx = n >> 10, gy = (n >> 5) & 31, gz = n & 31;
    float r0 = c0, r1 = c1;
    if (s < 2) {
      float ss = c0 * c0 + c1 * c1;
#pragma unroll
      for (int msk = 1; msk < 32; msk <<= 1) ss += __shfl_xor(ss, msk, 32);
      float nrm = fmaxf(sqrtf(ss), 1e-12f);
      float t0 = c0 / nrm * gam[h * 64 + 2 * lane] * 8.0f;
      float t1 = c1 / nrm * gam[h * 64 + 2 * lane + 1] * 8.0f;
      float ph = 0.0f;
      if (lane < 30) {
        int comp = lane / FREQD, fi = lane - comp * FREQD;
        int crd = (comp == 0) ? gx : ((comp == 1) ? gy : gz);
        ph = (float)crd * __expf(-(float)fi * (9.2103403720f / FREQD)); // ln(10000)
      }
      float cs = cosf(ph), sn = sinf(ph);
      r0 = t0 * cs - t1 * sn;
      r1 = t0 * sn + t1 * cs;
    }
    const int wdw = ((gx >> 3) * 4 + (gy >> 3)) * 4 + (gz >> 3);
    const int tt  = ((gx & 7) * 8 + (gy & 7)) * 8 + (gz & 7);
    const size_t wh = (size_t)(wdw * NH + h);
    if (s == 2) {            // V stored [d][t] so attention V tiles are 2D copies
      vwT[(wh * HD + 2 * lane) * TWIN + tt]     = (bf16)r0;
      vwT[(wh * HD + 2 * lane + 1) * TWIN + tt] = (bf16)r1;
    } else {
      bf16* dst = (s == 0 ? qw2 : kw2) + (wh * TWIN + tt) * HD + 2 * lane;
      v2bf o; o[0] = (bf16)r0; o[1] = (bf16)r1;
      *(v2bf*)dst = o;
    }
  }
}

// ------------------------------------------------ windowed flash attention
// grid: (4 q-chunks of 128, 16 heads, 64 windows), 256 threads = 8 waves x 16 rows
// K and V tiles double-buffered via the TDM; output in natural token order.
__global__ __launch_bounds__(256) void attn_win_kernel(
    const bf16* __restrict__ qw2, const bf16* __restrict__ kw2,
    const bf16* __restrict__ vwT, bf16* __restrict__ hwb) {
  __shared__ __align__(16) bf16 kBuf[2][32 * 64];   // [key][d]
  __shared__ __align__(16) bf16 vBuf[2][64 * 32];   // [d][key]
  __shared__ __align__(16) bf16 pLds[8 * 16 * 32];  // per-wave P tile

  const int w = blockIdx.z, h = blockIdx.y;
  const int tid = threadIdx.x, wave = tid >> 5, lane = tid & 31;
  const int qbase = blockIdx.x * 128 + wave * 16;
  const int kb8 = (lane >> 4) * 8;
  const int kb16 = (lane >> 4) * 16;
  const size_t wh = (size_t)(w * NH + h);
  const bf16* kSrc = kw2 + wh * TWIN * HD;   // + kc*HD, contiguous keys
  const bf16* vSrc = vwT + wh * HD * TWIN;   // + kc, row stride TWIN
  const int NITER = TWIN / 32;

  const int qrow = qbase + (lane & 15);
  const bf16* qp = qw2 + (wh * TWIN + qrow) * HD;
  v16bf qf0 = ld16(qp + kb16,      qp + kb16 + 8);       // d 0..31
  v16bf qf1 = ld16(qp + 32 + kb16, qp + 32 + kb16 + 8);  // d 32..63

  float mrow[8], lrow[8];
  v8f oacc[4];
#pragma unroll
  for (int r = 0; r < 8; ++r) { mrow[r] = -1e30f; lrow[r] = 0.0f; }
#pragma unroll
  for (int dt = 0; dt < 4; ++dt) zero8(oacc[dt]);

#ifdef HAVE_TDM
  if (tid == 0) {
    tdm_load_2d(lds_off_of(&kBuf[0][0]), (unsigned long long)(uintptr_t)kSrc, 64, 32, HD);
    tdm_load_2d(lds_off_of(&vBuf[0][0]), (unsigned long long)(uintptr_t)vSrc, 32, 64, TWIN);
    __builtin_amdgcn_s_wait_tensorcnt((short)0);
  }
#else
  {
    int key = tid >> 3, d8 = (tid & 7) * 8;
    *(v8bf*)&kBuf[0][key * 64 + d8] = *(const v8bf*)(kSrc + key * 64 + d8);
    int d = tid >> 2, q8 = (tid & 3) * 8;
    *(v8bf*)&vBuf[0][d * 32 + q8] = *(const v8bf*)(vSrc + (size_t)d * TWIN + q8);
  }
#endif
  __syncthreads();

  for (int it = 0; it < NITER; ++it) {
    const int cur = it & 1, nxt = cur ^ 1;
    const int kc = (it + 1) * 32;
    if (it + 1 < NITER) {   // stage next K/V while computing this chunk
#ifdef HAVE_TDM
      if (tid == 0) {
        tdm_load_2d(lds_off_of(&kBuf[nxt][0]),
                    (unsigned long long)(uintptr_t)(kSrc + (size_t)kc * HD), 64, 32, HD);
        tdm_load_2d(lds_off_of(&vBuf[nxt][0]),
                    (unsigned long long)(uintptr_t)(vSrc + kc), 32, 64, TWIN);
      }
#else
      int key = tid >> 3, d8 = (tid & 7) * 8;
      *(v8bf*)&kBuf[nxt][key * 64 + d8] = *(const v8bf*)(kSrc + (size_t)(kc + key) * 64 + d8);
      int d = tid >> 2, q8 = (tid & 3) * 8;
      *(v8bf*)&vBuf[nxt][d * 32 + q8] = *(const v8bf*)(vSrc + (size_t)d * TWIN + kc + q8);
#endif
    }
    const bf16* kC = kBuf[cur];
    const bf16* vC = vBuf[cur];

    v16bf kf[2][2];
#pragma unroll
    for (int nt = 0; nt < 2; ++nt) {
      const bf16* kp = &kC[(nt * 16 + (lane & 15)) * 64 + kb16];
      kf[nt][0] = *(const v16bf*)kp;
      kf[nt][1] = *(const v16bf*)(kp + 32);
    }
    v8f sacc[2];
    zero8(sacc[0]); zero8(sacc[1]);
    sacc[0] = __builtin_amdgcn_wmma_f32_16x16x32_bf16(false, qf0, false, kf[0][0],
                                                      (short)0, sacc[0], false, false);
    sacc[1] = __builtin_amdgcn_wmma_f32_16x16x32_bf16(false, qf0, false, kf[1][0],
                                                      (short)0, sacc[1], false, false);
    sacc[0] = __builtin_amdgcn_wmma_f32_16x16x32_bf16(false, qf1, false, kf[0][1],
                                                      (short)0, sacc[0], false, false);
    sacc[1] = __builtin_amdgcn_wmma_f32_16x16x32_bf16(false, qf1, false, kf[1][1],
                                                      (short)0, sacc[1], false, false);

    // online softmax; row r+8*(lane/16) lives in 16-lane halves -> xor masks 1..8
#pragma unroll
    for (int r = 0; r < 8; ++r) {
      float s0 = sacc[0][r] * SM_SCALE;
      float s1 = sacc[1][r] * SM_SCALE;
      float cur_m = fmaxf(s0, s1);
#pragma unroll
      for (int msk = 1; msk < 16; msk <<= 1) cur_m = fmaxf(cur_m, __shfl_xor(cur_m, msk, 32));
      float mnew = fmaxf(mrow[r], cur_m);
      float alpha = __expf(mrow[r] - mnew);
      float p0 = __expf(s0 - mnew);
      float p1 = __expf(s1 - mnew);
      float ps = p0 + p1;
#pragma unroll
      for (int msk = 1; msk < 16; msk <<= 1) ps += __shfl_xor(ps, msk, 32);
      lrow[r] = lrow[r] * alpha + ps;
      mrow[r] = mnew;
#pragma unroll
      for (int dt = 0; dt < 4; ++dt) oacc[dt][r] *= alpha;
      const int prow = r + 8 * (lane >> 4);
      pLds[wave * 512 + prow * 32 + (lane & 15)]      = (bf16)p0;
      pLds[wave * 512 + prow * 32 + 16 + (lane & 15)] = (bf16)p1;
    }
    // reload P as an A fragment (C/D layout -> A layout via LDS, intra-wave)
    const bf16* pp = &pLds[wave * 512 + (lane & 15) * 32];
    v16bf pf = ld16(pp + kb16, pp + kb16 + 8);
    v16bf vf[4];
#pragma unroll
    for (int dt = 0; dt < 4; ++dt)
      vf[dt] = *(const v16bf*)&vC[(dt * 16 + (lane & 15)) * 32 + kb16];
#pragma unroll
    for (int dt = 0; dt < 4; ++dt)
      oacc[dt] = __builtin_amdgcn_wmma_f32_16x16x32_bf16(false, pf, false, vf[dt],
                                                         (short)0, oacc[dt], false, false);
#ifdef HAVE_TDM
    if (tid == 0) __builtin_amdgcn_s_wait_tensorcnt((short)0);
#endif
    __syncthreads();
  }
  // epilogue: normalize and store in natural token order (inverse window permute)
  const int wx = w >> 4, wy = (w >> 2) & 3, wz = w & 3;
#pragma unroll
  for (int dt = 0; dt < 4; ++dt)
#pragma unroll
    for (int r = 0; r < 8; ++r) {
      const int row = qbase + r + 8 * (lane >> 4);
      const int lx = row >> 6, ly = (row >> 3) & 7, lz = row & 7;
      const int n = ((wx * 8 + lx) << 10) | ((wy * 8 + ly) << 5) | (wz * 8 + lz);
      hwb[(size_t)n * CD + h * HD + dt * 16 + (lane & 15)] = (bf16)(oacc[dt][r] / lrow[r]);
    }
}

// ------------------------------------------------ output projection GEMM
// grid: (16 column blocks of 64, 256 row tiles of 128), 256 threads
__global__ __launch_bounds__(256) void out_gemm_kernel(
    const bf16* __restrict__ hwb, const bf16* __restrict__ wobT,
    const float* __restrict__ bout, float* __restrict__ out) {
  __shared__ __align__(16) bf16 aBuf[2][128 * 64];
  __shared__ __align__(16) bf16 bBuf[2][64 * 64];
  const int ncol = blockIdx.x * 64;
  const int rowbase = blockIdx.y * 128;
  const int tid = threadIdx.x, wave = tid >> 5, lane = tid & 31;
  const int NITER = CD / 64;
  const bf16* aSrc = hwb + (size_t)rowbase * CD;
  const bf16* bSrc = wobT + (size_t)ncol * CD;

  v8f acc[4];
#pragma unroll
  for (int i = 0; i < 4; ++i) zero8(acc[i]);

#ifdef HAVE_TDM
  if (tid == 0) {
    tdm_load_2d(lds_off_of(&aBuf[0][0]), (unsigned long long)(uintptr_t)aSrc, 64, 128, CD);
    tdm_load_2d(lds_off_of(&bBuf[0][0]), (unsigned long long)(uintptr_t)bSrc, 64, 64, CD);
    __builtin_amdgcn_s_wait_tensorcnt((short)0);
  }
#else
  {
    int row = tid >> 1, half = tid & 1;
#pragma unroll
    for (int j = 0; j < 4; ++j)
      *(v8bf*)&aBuf[0][row * 64 + half * 32 + 8 * j] =
          *(const v8bf*)(aSrc + (size_t)row * CD + half * 32 + 8 * j);
    int n = tid >> 2, q = (tid & 3) * 16;
    *(v8bf*)&bBuf[0][n * 64 + q]     = *(const v8bf*)(bSrc + (size_t)n * CD + q);
    *(v8bf*)&bBuf[0][n * 64 + q + 8] = *(const v8bf*)(bSrc + (size_t)n * CD + q + 8);
  }
#endif
  __syncthreads();

  for (int it = 0; it < NITER; ++it) {
    const int cur = it & 1, nxt = cur ^ 1;
    const int kk = (it + 1) * 64;
    if (it + 1 < NITER) {
#ifdef HAVE_TDM
      if (tid == 0) {
        tdm_load_2d(lds_off_of(&aBuf[nxt][0]),
                    (unsigned long long)(uintptr_t)(aSrc + kk), 64, 128, CD);
        tdm_load_2d(lds_off_of(&bBuf[nxt][0]),
                    (unsigned long long)(uintptr_t)(bSrc + kk), 64, 64, CD);
      }
#else
      int row = tid >> 1, half = tid & 1;
#pragma unroll
      for (int j = 0; j < 4; ++j)
        *(v8bf*)&aBuf[nxt][row * 64 + half * 32 + 8 * j] =
            *(const v8bf*)(aSrc + (size_t)row * CD + kk + half * 32 + 8 * j);
      int n = tid >> 2, q = (tid & 3) * 16;
      *(v8bf*)&bBuf[nxt][n * 64 + q]     = *(const v8bf*)(bSrc + (size_t)n * CD + kk + q);
      *(v8bf*)&bBuf[nxt][n * 64 + q + 8] = *(const v8bf*)(bSrc + (size_t)n * CD + kk + q + 8);
#endif
    }
    const bf16* aC = aBuf[cur];
    const bf16* bC = bBuf[cur];
    const int m = wave * 16 + (lane & 15);
    const int kb8 = (lane >> 4) * 8;
    const int kb16 = (lane >> 4) * 16;
    v16bf af0 = ld16(&aC[m * 64 + kb16],      &aC[m * 64 + kb16 + 8]);
    v16bf af1 = ld16(&aC[m * 64 + 32 + kb16], &aC[m * 64 + 32 + kb16 + 8]);
    v16bf bf0[4], bf1[4];
#pragma unroll
    for (int nt = 0; nt < 4; ++nt) {
      const bf16* bp = &bC[(nt * 16 + (lane & 15)) * 64 + kb16];
      bf0[nt] = *(const v16bf*)bp;
      bf1[nt] = *(const v16bf*)(bp + 32);
    }
#pragma unroll
    for (int nt = 0; nt < 4; ++nt)
      acc[nt] = __builtin_amdgcn_wmma_f32_16x16x32_bf16(false, af0, false, bf0[nt],
                                                        (short)0, acc[nt], false, false);
#pragma unroll
    for (int nt = 0; nt < 4; ++nt)
      acc[nt] = __builtin_amdgcn_wmma_f32_16x16x32_bf16(false, af1, false, bf1[nt],
                                                        (short)0, acc[nt], false, false);
#ifdef HAVE_TDM
    if (tid == 0) __builtin_amdgcn_s_wait_tensorcnt((short)0);
#endif
    __syncthreads();
  }
#pragma unroll
  for (int nt = 0; nt < 4; ++nt)
#pragma unroll
    for (int r = 0; r < 8; ++r) {
      const int row = rowbase + wave * 16 + r + 8 * (lane >> 4);
      const int col = ncol + nt * 16 + (lane & 15);
      out[(size_t)row * CD + col] = acc[nt][r] + bout[col];
    }
}

// --------------------------------------------------------------- launcher
extern "C" void kernel_launch(void* const* d_in, const int* in_sizes, int n_in,
                              void* d_out, int out_size, void* d_ws, size_t ws_size,
                              hipStream_t stream) {
  const float* x    = (const float*)d_in[0];
  // d_in[1] = coords (recomputed on device from the linear index)
  const float* Wqkv = (const float*)d_in[2];
  const float* bqkv = (const float*)d_in[3];
  const float* gq   = (const float*)d_in[4];
  const float* gk   = (const float*)d_in[5];
  const float* Wout = (const float*)d_in[6];
  const float* bout = (const float*)d_in[7];
  float* out = (float*)d_out;

  char* ws = (char*)d_ws;
  size_t o = 0;
  bf16* xb   = (bf16*)(ws + o); o += (size_t)NTOK * CD * 2;
  bf16* wqbT = (bf16*)(ws + o); o += (size_t)CD * 3 * CD * 2;  // [3C][C]
  bf16* wobT = (bf16*)(ws + o); o += (size_t)CD * CD * 2;      // [C][C] n-major
  bf16* qw2  = (bf16*)(ws + o); o += (size_t)NTOK * CD * 2;    // [(w,h)][t][d]
  bf16* kw2  = (bf16*)(ws + o); o += (size_t)NTOK * CD * 2;    // [(w,h)][t][d]
  bf16* vwT  = (bf16*)(ws + o); o += (size_t)NTOK * CD * 2;    // [(w,h)][d][t]
  bf16* hwb  = (bf16*)(ws + o); o += (size_t)NTOK * CD * 2;    // natural order

  cvt_bf16_kernel<<<(NTOK * CD / 4 + 255) / 256, 256, 0, stream>>>(x, xb, NTOK * CD);
  cvt_transpose_kernel<<<(3 * CD * (CD / 8) + 255) / 256, 256, 0, stream>>>(
      Wqkv, wqbT, CD, 3 * CD);
  cvt_transpose_kernel<<<(CD * (CD / 8) + 255) / 256, 256, 0, stream>>>(
      Wout, wobT, CD, CD);

  qkv_gemm_kernel<<<dim3(48, NTOK / 128), 256, 0, stream>>>(xb, wqbT, bqkv, gq, gk,
                                                            qw2, kw2, vwT);
  attn_win_kernel<<<dim3(TWIN / 128, NH, NWIN), 256, 0, stream>>>(qw2, kw2, vwT, hwb);
  out_gemm_kernel<<<dim3(CD / 64, NTOK / 128), 256, 0, stream>>>(hwb, wobT, bout, out);
}
